// NNMFLinear_31456340476335
// MI455X (gfx1250) — compile-verified, matrix operand order checked by reading, not run on hardware
//
#include <hip/hip_runtime.h>
#include <hip/hip_bf16.h>

// NNMF multiplicative-update forward for MI455X (gfx1250, wave32, WMMA).
// Working set (~140 MB) stays L2-resident (192 MB); both GEMMs per iteration
// run on V_WMMA_F32_16X16X32_F16 (f16 in, f32 accumulate). Tile staging uses
// the CDNA5 async global->LDS path (ASYNCcnt) when the toolchain exposes it;
// epilogues are fully coalesced via an LDS accumulator transpose.

typedef __attribute__((ext_vector_type(16))) _Float16 v16h;
typedef __attribute__((ext_vector_type(8)))  _Float16 v8h;
typedef __attribute__((ext_vector_type(8)))  float    v8f;
typedef __attribute__((ext_vector_type(4)))  float    v4f;
typedef __attribute__((ext_vector_type(4)))  int      v4i;

#if __has_builtin(__builtin_amdgcn_global_load_async_to_lds_b128) && \
    __has_builtin(__builtin_amdgcn_s_wait_asynccnt)
#define USE_ASYNC_LDS 1
typedef __attribute__((address_space(1))) v4i* gas_b128_t;  // global int4*
typedef __attribute__((address_space(3))) v4i* lds_b128_t;  // LDS int4*
#else
#define USE_ASYNC_LDS 0
#endif

#define B_ROWS 8192
#define NIN    3072
#define NOUT   512

#define BM 64
#define BN 256
#define BK 32
#define SA (BK + 8)    // padded LDS row stride (halfs) for staging tiles
#define SB (BK + 8)
#define CS (BN + 4)    // padded LDS row stride (floats) for C transpose

// ---------------------------------------------------------------------------
// x row-normalization: x_norm[b,j] = x[b,j] / (sum_j x[b,j] + eps), stored f16
// ---------------------------------------------------------------------------
__launch_bounds__(256)
__global__ void nnmf_xnorm_kernel(const float* __restrict__ x,
                                  _Float16* __restrict__ xnorm) {
    __shared__ float red[256];
    const int b = blockIdx.x;
    const int t = threadIdx.x;
    const float* xr = x + (long)b * NIN;
    float s = 0.0f;
    for (int i = t; i < NIN; i += 256) s += xr[i];
    red[t] = s;
    __syncthreads();
    for (int off = 128; off > 0; off >>= 1) {
        if (t < off) red[t] += red[t + off];
        __syncthreads();
    }
    const float inv = 1.0f / (red[0] + 1e-20f);
    _Float16* out = xnorm + (long)b * NIN;
    for (int i = t; i < NIN; i += 256) out[i] = (_Float16)(xr[i] * inv);
}

// ---------------------------------------------------------------------------
// W f32 -> f16 row-major (B operand of GEMM2) and f16 transposed (B of GEMM1)
// ---------------------------------------------------------------------------
__launch_bounds__(256)
__global__ void nnmf_wconv_kernel(const float* __restrict__ w,
                                  _Float16* __restrict__ wf16,
                                  _Float16* __restrict__ wtf16) {
    const long i = (long)blockIdx.x * 256 + threadIdx.x;  // over NOUT*NIN
    const int n = (int)(i / NIN);
    const int j = (int)(i % NIN);
    const float v = w[i];
    wf16[i] = (_Float16)v;                       // [NOUT, NIN]
    wtf16[(long)j * NOUT + n] = (_Float16)v;     // [NIN, NOUT]
}

// ---------------------------------------------------------------------------
// h init: broadcast h_initial across all rows (f32 state + f16 GEMM operand)
// ---------------------------------------------------------------------------
__launch_bounds__(256)
__global__ void nnmf_inith_kernel(const float* __restrict__ h0,
                                  float* __restrict__ h_f32,
                                  _Float16* __restrict__ h_f16) {
    const long i = (long)blockIdx.x * 256 + threadIdx.x;  // over B_ROWS*NOUT
    const float v = h0[i & (NOUT - 1)];
    h_f32[i] = v;
    h_f16[i] = (_Float16)v;
}

// ---------------------------------------------------------------------------
// Unified WMMA GEMM: C[M,N] = A[M,K] @ B  (B supplied "column major": B[n,k])
//   MODE 0: GEMM1  (A=h_f16, B=Wt)  epilogue: ratio = x_norm/(acc+eps) -> f16
//   MODE 1: GEMM2  (A=ratio, B=W)   epilogue: s_out = acc (f32)
// Block tile 64x256, 8 waves as 2(row) x 4(col), each wave 2x4 16x16 accums.
// Staging: async global->LDS (no VGPR round-trip) when available.
// Epilogue transposes accumulators through LDS for coalesced 16B global I/O.
// ---------------------------------------------------------------------------
template <int MODE>
__launch_bounds__(256)
__global__ void nnmf_gemm_kernel(const _Float16* __restrict__ A, int lda,
                                 const _Float16* __restrict__ Bc, int ldb,
                                 int N, int K,
                                 const _Float16* __restrict__ xnorm,
                                 _Float16* __restrict__ ratio_out,
                                 float* __restrict__ s_out) {
    // Staging tiles live in the same LDS bytes as the epilogue's C transpose
    // (they are dead by the time the epilogue runs; a barrier separates them).
    __shared__ __align__(16) char shraw[BM * CS * 4];       // 66560 B
    _Float16* aSh = (_Float16*)shraw;                       // BM*SA halfs
    _Float16* bSh = (_Float16*)(shraw + BM * SA * 2);       // BN*SB halfs
    float*    cSh = (float*)shraw;                          // BM*CS floats

    const int tid  = threadIdx.x;
    const int lane = tid & 31;
    const int wave = tid >> 5;        // 0..7
    const int waveRow = wave >> 2;    // 0..1 (32 rows each)
    const int waveCol = wave & 3;     // 0..3 (64 cols each)

    const int blockM = blockIdx.x * BM;
    const int blockN = blockIdx.y * BN;

    // ISA 7.12.2 per-lane fragment pattern (16-bit A/B, 16x32 / 32x16):
    //   lanes 0..15 : row/col = lane,    K = {0..7, 16..23}
    //   lanes 16..31: row/col = lane-16, K = {8..15, 24..31}
    const int lh  = lane & 15;
    const int sel = (lane >> 4) << 3;   // 0 or 8 halfs

    v8f zero = {0.f, 0.f, 0.f, 0.f, 0.f, 0.f, 0.f, 0.f};
    v8f acc[2][4];
#pragma unroll
    for (int rt = 0; rt < 2; ++rt)
#pragma unroll
        for (int ct = 0; ct < 4; ++ct) acc[rt][ct] = zero;

    for (int k0 = 0; k0 < K; k0 += BK) {
        // ---- stage A tile (64 x 32 halfs): 256 chunks of 16B, one per thread
        {
            const int m  = tid >> 2;
            const int kk = (tid & 3) << 3;
            const _Float16* gsrc = &A[(long)(blockM + m) * lda + k0 + kk];
            _Float16* ldst = &aSh[m * SA + kk];
#if USE_ASYNC_LDS
            __builtin_amdgcn_global_load_async_to_lds_b128(
                (gas_b128_t)gsrc, (lds_b128_t)ldst, 0, 0);
#else
            *(v8h*)ldst = *(const v8h*)gsrc;
#endif
        }
        // ---- stage B tile (256 x 32 halfs): 1024 chunks, 4 per thread
#pragma unroll
        for (int c = 0; c < 4; ++c) {
            const int idx = tid + c * 256;
            const int n   = idx >> 2;
            const int kk  = (idx & 3) << 3;
            const _Float16* gsrc = &Bc[(long)(blockN + n) * ldb + k0 + kk];
            _Float16* ldst = &bSh[n * SB + kk];
#if USE_ASYNC_LDS
            __builtin_amdgcn_global_load_async_to_lds_b128(
                (gas_b128_t)gsrc, (lds_b128_t)ldst, 0, 0);
#else
            *(v8h*)ldst = *(const v8h*)gsrc;
#endif
        }
#if USE_ASYNC_LDS
        __builtin_amdgcn_s_wait_asynccnt(0);
#endif
        __syncthreads();

        // ---- build fragments from LDS and issue WMMAs
        v16h aF[2], bF[4];
#pragma unroll
        for (int rt = 0; rt < 2; ++rt) {
            const int row = waveRow * 32 + rt * 16 + lh;
            v8h lo = *(const v8h*)&aSh[row * SA + sel];
            v8h hi = *(const v8h*)&aSh[row * SA + 16 + sel];
            aF[rt] = __builtin_shufflevector(lo, hi, 0, 1, 2, 3, 4, 5, 6, 7,
                                             8, 9, 10, 11, 12, 13, 14, 15);
        }
#pragma unroll
        for (int ct = 0; ct < 4; ++ct) {
            const int col = waveCol * 64 + ct * 16 + lh;
            v8h lo = *(const v8h*)&bSh[col * SB + sel];
            v8h hi = *(const v8h*)&bSh[col * SB + 16 + sel];
            bF[ct] = __builtin_shufflevector(lo, hi, 0, 1, 2, 3, 4, 5, 6, 7,
                                             8, 9, 10, 11, 12, 13, 14, 15);
        }
#pragma unroll
        for (int rt = 0; rt < 2; ++rt)
#pragma unroll
            for (int ct = 0; ct < 4; ++ct)
                acc[rt][ct] = __builtin_amdgcn_wmma_f32_16x16x32_f16(
                    false, aF[rt], false, bF[ct], (short)0, acc[rt][ct],
                    false, false);
        __syncthreads();
    }

    // ---- epilogue: deposit accumulators into LDS (C/D layout: lanes 0..15 ->
    //      N=lane, M=vgpr; lanes 16..31 -> N=lane-16, M=vgpr+8), then stream
    //      the 64x256 tile out row-major with 16B coalesced global accesses.
    const int nIdx = lane & 15;
    const int mOff = (lane >> 4) << 3;
#pragma unroll
    for (int rt = 0; rt < 2; ++rt)
#pragma unroll
        for (int ct = 0; ct < 4; ++ct) {
            const int ml = waveRow * 32 + rt * 16 + mOff;
            const int nl = waveCol * 64 + ct * 16 + nIdx;
#pragma unroll
            for (int v = 0; v < 8; ++v)
                cSh[(ml + v) * CS + nl] = acc[rt][ct][v];
        }
    __syncthreads();

    // 64 rows x 256 cols = 2048 chunks of 8 elems; 8 chunks per thread.
#pragma unroll
    for (int c = 0; c < 8; ++c) {
        const int idx = tid + c * 256;
        const int m   = idx >> 5;          // 32 chunks per row
        const int nn  = (idx & 31) << 3;
        const long g  = (long)(blockM + m) * N + blockN + nn;
        const v4f clo = *(const v4f*)&cSh[m * CS + nn];
        const v4f chi = *(const v4f*)&cSh[m * CS + nn + 4];
        if (MODE == 0) {
            const v8h xv = *(const v8h*)&xnorm[g];
            v8h rv;
#pragma unroll
            for (int e = 0; e < 4; ++e) {
                // v_rcp_f32 (~1 ulp) is exact enough for an f16-rounded result
                rv[e]     = (_Float16)((float)xv[e] *
                                       __builtin_amdgcn_rcpf(clo[e] + 1e-20f));
                rv[e + 4] = (_Float16)((float)xv[e + 4] *
                                       __builtin_amdgcn_rcpf(chi[e] + 1e-20f));
            }
            *(v8h*)&ratio_out[g] = rv;
        } else {
            *(v4f*)&s_out[g]     = clo;
            *(v4f*)&s_out[g + 4] = chi;
        }
    }
}

// ---------------------------------------------------------------------------
// Row update + normalize: hnew = h*(1+s); hnew /= (rowsum + eps).
// One block per row (NOUT=512 -> 2 elements/thread).
// ---------------------------------------------------------------------------
__launch_bounds__(256)
__global__ void nnmf_norm_kernel(const float* __restrict__ h_in,
                                 const float* __restrict__ s,
                                 float* __restrict__ h_out,
                                 _Float16* __restrict__ h_out_f16) {
    __shared__ float red[256];
    const int b = blockIdx.x;
    const int t = threadIdx.x;
    const long base = (long)b * NOUT;
    const float v0 = h_in[base + t]       * (1.0f + s[base + t]);
    const float v1 = h_in[base + t + 256] * (1.0f + s[base + t + 256]);
    red[t] = v0 + v1;
    __syncthreads();
    for (int off = 128; off > 0; off >>= 1) {
        if (t < off) red[t] += red[t + off];
        __syncthreads();
    }
    const float inv = 1.0f / (red[0] + 1e-20f);
    const float o0 = v0 * inv, o1 = v1 * inv;
    h_out[base + t]           = o0;
    h_out[base + t + 256]     = o1;
    h_out_f16[base + t]       = (_Float16)o0;
    h_out_f16[base + t + 256] = (_Float16)o1;
}

// ---------------------------------------------------------------------------
extern "C" void kernel_launch(void* const* d_in, const int* in_sizes, int n_in,
                              void* d_out, int out_size, void* d_ws, size_t ws_size,
                              hipStream_t stream) {
    const float* x  = (const float*)d_in[0];  // [8192, 3072]
    const float* w  = (const float*)d_in[1];  // [512, 3072]
    const float* h0 = (const float*)d_in[2];  // [512]
    float* out = (float*)d_out;               // [8192, 512]

    // workspace layout (all offsets 256B aligned)
    char* ws = (char*)d_ws;
    _Float16* xnorm = (_Float16*)(ws + 0);                       // 50331648 B
    _Float16* ratio = (_Float16*)(ws + 50331648L);               // 50331648 B
    _Float16* wf16  = (_Float16*)(ws + 100663296L);              //  3145728 B
    _Float16* wtf16 = (_Float16*)(ws + 103809024L);              //  3145728 B
    float*    hf32  = (float*)   (ws + 106954752L);              // 16777216 B
    _Float16* hf16  = (_Float16*)(ws + 123731968L);              //  8388608 B
    float*    sbuf  = (float*)   (ws + 132120576L);              // 16777216 B

    // prep (re-run every call for determinism)
    nnmf_xnorm_kernel<<<B_ROWS, 256, 0, stream>>>(x, xnorm);
    nnmf_wconv_kernel<<<(NOUT * NIN) / 256, 256, 0, stream>>>(w, wf16, wtf16);
    nnmf_inith_kernel<<<(B_ROWS * NOUT) / 256, 256, 0, stream>>>(h0, hf32, hf16);

    const dim3 g1(B_ROWS / BM, NIN / BN);   // 128 x 12 blocks
    const dim3 g2(B_ROWS / BM, NOUT / BN);  // 128 x 2 blocks

    for (int it = 0; it < 5; ++it) {
        // GEMM1: denom = h @ W ; fused ratio = x_norm/(denom+eps) -> f16
        nnmf_gemm_kernel<0><<<g1, 256, 0, stream>>>(
            hf16, NOUT, wtf16, NOUT, NIN, NOUT, xnorm, ratio, nullptr);
        // GEMM2: s = ratio @ W^T -> f32
        nnmf_gemm_kernel<1><<<g2, 256, 0, stream>>>(
            ratio, NIN, wf16, NIN, NOUT, NIN, nullptr, nullptr, sbuf);
        // h <- normalize(h * (1 + s)); final iteration writes d_out
        float* hdst = (it == 4) ? out : hf32;
        nnmf_norm_kernel<<<B_ROWS, 256, 0, stream>>>(hf32, sbuf, hdst, hf16);
    }
}
